// DocGCN_35175782154375
// MI455X (gfx1250) — matrix operand order repaired and checked by reading, not compiled, not run on hardware
//
#include <hip/hip_runtime.h>
#include <hip/hip_bf16.h>
#include <math.h>

// ---------------- constants ----------------
#define CB 64
#define CL 512
#define CH 768
#define CT 256
#define CS 128
#define CD 8
#define CN (CB * CS)          // 8192 nodes
#define NEG_SLOPE 0.2f
#define NEGINF_KEY 0x007FFFFFu   // fkey(-inf)

typedef __attribute__((ext_vector_type(16))) _Float16 v16h;
typedef __attribute__((ext_vector_type(8)))  _Float16 v8h;
typedef __attribute__((ext_vector_type(8)))  float    v8f;

// monotonic unsigned encoding of float for atomicMax
__device__ __forceinline__ unsigned fkey(float f) {
    unsigned u = __float_as_uint(f);
    return (u & 0x80000000u) ? ~u : (u | 0x80000000u);
}
__device__ __forceinline__ float fdec(unsigned k) {
    unsigned u = (k & 0x80000000u) ? (k ^ 0x80000000u) : ~k;
    return __uint_as_float(u);
}

// ---------------- kernel 1: span pooling + gather -> h ----------------
__global__ __launch_bounds__(256)
void k_build_h(const float* __restrict__ feat, const int* __restrict__ spans,
               const int* __restrict__ masks, const int* __restrict__ sel,
               float* __restrict__ h32, _Float16* __restrict__ h16)
{
    int n = blockIdx.x;                 // node
    int b = n / CS, si = n % CS;
    int t = sel[b * CS + si];
    int s = spans[(b * CT + t) * 2 + 0];
    int e = spans[(b * CT + t) * 2 + 1];
    int lo = s < 0 ? 0 : s;
    int hi = e > CL ? CL : e;
    int cnt = 0;
    for (int l = lo; l < hi; ++l) if (masks[b * CL + l] > 0) cnt++;
    float inv = 1.0f / (float)(cnt > 0 ? cnt : 1);
    float zf  = (e > 0) ? 1.0f : 0.0f;
    for (int hh = threadIdx.x; hh < CH; hh += 256) {
        float acc = 0.f;
        for (int l = lo; l < hi; ++l)
            if (masks[b * CL + l] > 0)
                acc += feat[((size_t)b * CL + l) * CH + hh];
        float v = acc * inv * zf;
        h32[(size_t)n * CH + hh] = v;
        h16[(size_t)n * CH + hh] = (_Float16)v;
    }
}

// ---------------- weight convert + transpose: Wt[n][k] = (f16)W[k][n] ----------------
__global__ __launch_bounds__(256)
void k_convert_w(const float* __restrict__ W, _Float16* __restrict__ Wt)
{
    int i = blockIdx.x * 256 + threadIdx.x;       // < CH*CH
    int k = i / CH, n = i % CH;
    Wt[(size_t)n * CH + k] = (_Float16)W[i];
}

// ---------------- WMMA GEMM: z(N,H) = h16(N,H) @ W(H,H) ----------------
// Block = one 64-column strip of W  x  8 consecutive 16-row M tiles (8 waves).
// The W strip is staged cooperatively into LDS with GLOBAL_LOAD_ASYNC_TO_LDS_B128
// (ASYNCcnt path) in two 384-wide K halves; all 8 waves read B-frags from LDS.
#define KHALF 384
#define PADB  392            // 384 + 8 halves pad: 784B row stride -> bank-disjoint
__global__ __launch_bounds__(256)
void k_gemm_wmma(const _Float16* __restrict__ A,   // N x H row-major f16
                 const _Float16* __restrict__ Bt,  // H x H, Bt[n*H+k] = W[k*H+n]
                 float* __restrict__ C)            // N x H f32
{
    __shared__ _Float16 bsh[64 * PADB];            // 49,152+1024 B

    const int NSTRIP = (CH / 16) / 4;              // 12 strips of 64 cols
    int strip  = blockIdx.x % NSTRIP;
    int tm     = (blockIdx.x / NSTRIP) * 8 + (threadIdx.x >> 5);
    int lane   = threadIdx.x & 31;
    int g  = lane >> 4;
    int ml = lane & 15;

    const _Float16* arow = A + (size_t)(tm * 16 + ml) * CH;
    const _Float16* l0 = bsh + (size_t)(ml +  0) * PADB;
    const _Float16* l1 = bsh + (size_t)(ml + 16) * PADB;
    const _Float16* l2 = bsh + (size_t)(ml + 32) * PADB;
    const _Float16* l3 = bsh + (size_t)(ml + 48) * PADB;

    v8f c0 = {}, c1 = {}, c2 = {}, c3 = {};
    for (int kh = 0; kh < 2; ++kh) {
        // ---- async stage: 64 rows x 384 halves of the strip -> LDS ----
        {
            const _Float16* gbase = Bt + (size_t)strip * 64 * CH + kh * KHALF;
            #pragma unroll
            for (int i = 0; i < 12; ++i) {
                int c   = threadIdx.x + 256 * i;   // 0..3071 chunks of 16B
                int row = c / 48;
                int cj  = c % 48;
                unsigned ldsa = (unsigned)(size_t)(const void*)(bsh + row * PADB + cj * 8);
                const _Float16* gp = gbase + (size_t)row * CH + cj * 8;
                asm volatile("global_load_async_to_lds_b128 %0, %1, off"
                             :: "v"(ldsa), "v"(gp) : "memory");
            }
            asm volatile("s_wait_asynccnt 0x0" ::: "memory");
        }
        __syncthreads();

        // ---- 12 K-steps of 32 out of this half ----
        for (int k2 = 0; k2 < KHALF; k2 += 32) {
            int kk = kh * KHALF + k2;
            // A-frag: lane l holds row (l&15); K = kk+g*8..+7 (v0..3), kk+16+g*8..+7 (v4..7)
            v8h alo = *(const v8h*)(arow + kk + g * 8);
            v8h ahi = *(const v8h*)(arow + kk + 16 + g * 8);
            v16h a;
            #pragma unroll
            for (int i = 0; i < 8; ++i) { a[i] = alo[i]; a[8 + i] = ahi[i]; }
            // B-frag: lane l holds col (l&15); K = k2+g*8..+7 (v0..3), k2+16+g*8..+7 (v4..7) (LDS)
            int kb = k2 + g * 8;
            v8h t0a = *(const v8h*)(l0 + kb), t0b = *(const v8h*)(l0 + kb + 16);
            v8h t1a = *(const v8h*)(l1 + kb), t1b = *(const v8h*)(l1 + kb + 16);
            v8h t2a = *(const v8h*)(l2 + kb), t2b = *(const v8h*)(l2 + kb + 16);
            v8h t3a = *(const v8h*)(l3 + kb), t3b = *(const v8h*)(l3 + kb + 16);
            v16h b0, b1, b2, b3;
            #pragma unroll
            for (int i = 0; i < 8; ++i) {
                b0[i] = t0a[i]; b0[8 + i] = t0b[i];
                b1[i] = t1a[i]; b1[8 + i] = t1b[i];
                b2[i] = t2a[i]; b2[8 + i] = t2b[i];
                b3[i] = t3a[i]; b3[8 + i] = t3b[i];
            }
            c0 = __builtin_amdgcn_wmma_f32_16x16x32_f16(false, a, false, b0, (short)0, c0, false, false);
            c1 = __builtin_amdgcn_wmma_f32_16x16x32_f16(false, a, false, b1, (short)0, c1, false, false);
            c2 = __builtin_amdgcn_wmma_f32_16x16x32_f16(false, a, false, b2, (short)0, c2, false, false);
            c3 = __builtin_amdgcn_wmma_f32_16x16x32_f16(false, a, false, b3, (short)0, c3, false, false);
        }
        __syncthreads();   // protect LDS before next half overwrites
    }

    // C-frag: vgpr r -> row r + 8*g, col = lane&15
    float* crow = C + (size_t)(tm * 16 + g * 8) * CH + strip * 64 + ml;
    #pragma unroll
    for (int r = 0; r < 8; ++r) {
        crow[(size_t)r * CH +  0] = c0[r];
        crow[(size_t)r * CH + 16] = c1[r];
        crow[(size_t)r * CH + 32] = c2[r];
        crow[(size_t)r * CH + 48] = c3[r];
    }
}

// ---------------- el/er: per-row dot with al / ar ----------------
__global__ __launch_bounds__(256)
void k_attn_scores(const float* __restrict__ z, const float* __restrict__ al,
                   const float* __restrict__ ar, float* __restrict__ el,
                   float* __restrict__ er)
{
    __shared__ float sl[256], sr[256];
    int n = blockIdx.x;
    float pl = 0.f, pr = 0.f;
    for (int hh = threadIdx.x; hh < CH; hh += 256) {
        float v = z[(size_t)n * CH + hh];
        pl += v * al[hh];
        pr += v * ar[hh];
    }
    sl[threadIdx.x] = pl; sr[threadIdx.x] = pr; __syncthreads();
    for (int o = 128; o > 0; o >>= 1) {
        if (threadIdx.x < o) { sl[threadIdx.x] += sl[threadIdx.x + o];
                               sr[threadIdx.x] += sr[threadIdx.x + o]; }
        __syncthreads();
    }
    if (threadIdx.x == 0) { el[n] = sl[0]; er[n] = sr[0]; }
}

// ---------------- per-layer node init ----------------
__global__ void k_init_nodes(unsigned* __restrict__ maxkey, float* __restrict__ sumex)
{
    int i = blockIdx.x * 256 + threadIdx.x;
    if (i < CN) { maxkey[i] = NEGINF_KEY; sumex[i] = 0.f; }
}

__global__ void k_zero_f32(float* __restrict__ p, int n)
{
    for (int i = blockIdx.x * 256 + threadIdx.x; i < n; i += gridDim.x * 256) p[i] = 0.f;
}

// ---------------- edge pass 1: leaky-relu score + segment max ----------------
__global__ void k_edge_pass1(const int* __restrict__ src, const int* __restrict__ dst,
                             const float* __restrict__ el, const float* __restrict__ er,
                             float* __restrict__ ebuf, unsigned* __restrict__ maxkey, int E)
{
    int i = blockIdx.x * 256 + threadIdx.x;
    if (i >= E) return;
    float e = el[src[i]] + er[dst[i]];
    e = (e > 0.f) ? e : NEG_SLOPE * e;
    ebuf[i] = e;
    atomicMax(&maxkey[dst[i]], fkey(e));
}

// ---------------- edge pass 2: exp + segment sum ----------------
__global__ void k_edge_pass2(const int* __restrict__ dst, const float* __restrict__ ebuf,
                             const unsigned* __restrict__ maxkey, float* __restrict__ exbuf,
                             float* __restrict__ sumex, int E)
{
    int i = blockIdx.x * 256 + threadIdx.x;
    if (i >= E) return;
    int d = dst[i];
    float ex = expf(ebuf[i] - fdec(maxkey[d]));
    exbuf[i] = ex;
    atomicAdd(&sumex[d], ex);
}

// ---------------- edge pass 3: weighted scatter accum[dst] += a * z[src] ----------------
__global__ __launch_bounds__(256)
void k_edge_agg(const int* __restrict__ src, const int* __restrict__ dst,
                const float* __restrict__ exbuf, const float* __restrict__ sumex,
                const float* __restrict__ z, float* __restrict__ accum)
{
    int i = blockIdx.x;
    int s = src[i], d = dst[i];
    float coef = exbuf[i] / sumex[d];
    const float* zs = z + (size_t)s * CH;
    float* ad = accum + (size_t)d * CH;
    for (int hh = threadIdx.x; hh < CH; hh += 256)
        atomicAdd(&ad[hh], coef * zs[hh]);
}

// ---------------- elu finalize -> next-layer h (f32 + f16) ----------------
__global__ void k_finalize_elu(const float* __restrict__ accum, float* __restrict__ h32,
                               _Float16* __restrict__ h16, int n)
{
    int i = blockIdx.x * 256 + threadIdx.x;
    if (i >= n) return;
    float v = accum[i];
    float o = (v > 0.f) ? v : expm1f(v);
    h32[i] = o;
    h16[i] = (_Float16)o;
}

// ---------------- doc bookkeeping ----------------
__global__ void k_sentdoc(const int* __restrict__ doc_spans, int* __restrict__ sentdoc)
{
    int b = threadIdx.x;            // one block of 64
    if (b >= CB) return;
    int sd = -1;
    for (int d = 0; d < CD; ++d)
        if (doc_spans[2 * d] <= b) sd = d;
    sentdoc[b] = sd;
}

__global__ void k_nodecnt(const int* __restrict__ sentdoc, int* __restrict__ ncnt)
{
    int d = threadIdx.x;            // one block of 32
    if (d >= CD) return;
    int c = 0;
    for (int b = 0; b < CB; ++b)
        if (sentdoc[b] == d) c += CS;
    ncnt[d] = c;
}

// ---------------- qf: masked feature sum per doc ----------------
__global__ __launch_bounds__(256)
void k_qf(const float* __restrict__ feat, const int* __restrict__ seg,
          const int* __restrict__ ishead, const int* __restrict__ masks,
          const int* __restrict__ sentdoc, float* __restrict__ qfsum)
{
    __shared__ float qm[CL];
    int b = blockIdx.x;
    for (int l = threadIdx.x; l < CL; l += 256) {
        int ok = (ishead[b * CL + l] != 2) && (seg[b * CL + l] == 0) && (masks[b * CL + l] > 0);
        qm[l] = ok ? 1.f : 0.f;
    }
    __syncthreads();
    int d = sentdoc[b];
    for (int hh = threadIdx.x; hh < CH; hh += 256) {
        float acc = 0.f;
        for (int l = 0; l < CL; ++l)
            if (qm[l] != 0.f)
                acc += feat[((size_t)b * CL + l) * CH + hh];
        atomicAdd(&qfsum[d * CH + hh], acc);
    }
}

// ---------------- avg: per-doc node sums of h ----------------
__global__ __launch_bounds__(256)
void k_avg(const float* __restrict__ h32, const int* __restrict__ sentdoc,
           float* __restrict__ avgsum)
{
    int d = blockIdx.x;
    int hh = blockIdx.y * 256 + threadIdx.x;
    float acc = 0.f;
    for (int b = 0; b < CB; ++b) {
        if (sentdoc[b] != d) continue;
        const float* base = h32 + (size_t)b * CS * CH + hh;
        for (int r = 0; r < CS; ++r) acc += base[(size_t)r * CH];
    }
    avgsum[d * CH + hh] = acc;
}

// ---------------- final: out[d] = sum_h |qf - avg| ----------------
__global__ __launch_bounds__(256)
void k_final(const float* __restrict__ qfsum, const float* __restrict__ avgsum,
             const int* __restrict__ ncnt, const int* __restrict__ doc_spans,
             float* __restrict__ out)
{
    __shared__ float sm[256];
    int d = blockIdx.x;
    float dc = (float)(doc_spans[2 * d + 1] - doc_spans[2 * d]); if (dc < 1.f) dc = 1.f;
    float nc = (float)ncnt[d];                                    if (nc < 1.f) nc = 1.f;
    float acc = 0.f;
    for (int hh = threadIdx.x; hh < CH; hh += 256)
        acc += fabsf(qfsum[d * CH + hh] / dc - avgsum[d * CH + hh] / nc);
    sm[threadIdx.x] = acc; __syncthreads();
    for (int o = 128; o > 0; o >>= 1) {
        if (threadIdx.x < o) sm[threadIdx.x] += sm[threadIdx.x + o];
        __syncthreads();
    }
    if (threadIdx.x == 0) out[d] = sm[0];
}

// ---------------- host ----------------
static inline size_t align256(size_t x) { return (x + 255) & ~(size_t)255; }

extern "C" void kernel_launch(void* const* d_in, const int* in_sizes, int n_in,
                              void* d_out, int out_size, void* d_ws, size_t ws_size,
                              hipStream_t stream)
{
    (void)n_in; (void)out_size; (void)ws_size;
    const float* feat   = (const float*)d_in[0];
    const int*   spans  = (const int*)d_in[1];
    const int*   masks  = (const int*)d_in[2];
    const int*   sel    = (const int*)d_in[3];
    const int*   src    = (const int*)d_in[4];
    const int*   dst    = (const int*)d_in[5];
    const int*   dspan  = (const int*)d_in[6];
    const int*   seg    = (const int*)d_in[7];
    const int*   ishead = (const int*)d_in[8];
    const int Etot = in_sizes[4];

    char* ws = (char*)d_ws;
    size_t off = 0;
    const size_t NH4 = (size_t)CN * CH * 4, NH2 = (size_t)CN * CH * 2;
    float*      h32    = (float*)(ws + off);     off = align256(off + NH4);
    float*      z      = (float*)(ws + off);     off = align256(off + NH4);
    float*      accum  = (float*)(ws + off);     off = align256(off + NH4);
    _Float16*   h16    = (_Float16*)(ws + off);  off = align256(off + NH2);
    _Float16*   Wt     = (_Float16*)(ws + off);  off = align256(off + (size_t)CH * CH * 2);
    float*      el     = (float*)(ws + off);     off = align256(off + CN * 4);
    float*      er     = (float*)(ws + off);     off = align256(off + CN * 4);
    float*      ebuf   = (float*)(ws + off);     off = align256(off + (size_t)Etot * 4);
    float*      exbuf  = (float*)(ws + off);     off = align256(off + (size_t)Etot * 4);
    unsigned*   maxkey = (unsigned*)(ws + off);  off = align256(off + CN * 4);
    float*      sumex  = (float*)(ws + off);     off = align256(off + CN * 4);
    float*      qfsum  = (float*)(ws + off);     off = align256(off + CD * CH * 4);
    float*      avgsum = (float*)(ws + off);     off = align256(off + CD * CH * 4);
    int*        sentdoc= (int*)(ws + off);       off = align256(off + CB * 4);
    int*        ncnt   = (int*)(ws + off);       off = align256(off + CD * 4);

    // stage 1: node features + doc bookkeeping + query path
    k_build_h<<<CN, 256, 0, stream>>>(feat, spans, masks, sel, h32, h16);
    k_sentdoc<<<1, 64, 0, stream>>>(dspan, sentdoc);
    k_nodecnt<<<1, 32, 0, stream>>>(sentdoc, ncnt);
    k_zero_f32<<<24, 256, 0, stream>>>(qfsum, CD * CH);
    k_qf<<<CB, 256, 0, stream>>>(feat, seg, ishead, masks, sentdoc, qfsum);

    // stage 2: 3 GAT layers
    const int Eblocks = (Etot + 255) / 256;
    for (int layer = 0; layer < 3; ++layer) {
        const float* W  = (const float*)d_in[9 + 3 * layer];
        const float* al = (const float*)d_in[10 + 3 * layer];
        const float* ar = (const float*)d_in[11 + 3 * layer];
        k_convert_w<<<(CH * CH) / 256, 256, 0, stream>>>(W, Wt);
        k_gemm_wmma<<<(CN / 16) * ((CH / 16) / 4) / 8, 256, 0, stream>>>(h16, Wt, z);
        k_attn_scores<<<CN, 256, 0, stream>>>(z, al, ar, el, er);
        k_init_nodes<<<CN / 256, 256, 0, stream>>>(maxkey, sumex);
        k_zero_f32<<<4096, 256, 0, stream>>>(accum, CN * CH);
        k_edge_pass1<<<Eblocks, 256, 0, stream>>>(src, dst, el, er, ebuf, maxkey, Etot);
        k_edge_pass2<<<Eblocks, 256, 0, stream>>>(dst, ebuf, maxkey, exbuf, sumex, Etot);
        k_edge_agg<<<Etot, 256, 0, stream>>>(src, dst, exbuf, sumex, z, accum);
        k_finalize_elu<<<(CN * CH) / 256, 256, 0, stream>>>(accum, h32, h16, CN * CH);
    }

    // stage 3: doc reductions + output
    k_avg<<<dim3(CD, CH / 256), 256, 0, stream>>>(h32, sentdoc, avgsum);
    k_final<<<CD, 256, 0, stream>>>(qfsum, avgsum, ncnt, dspan, (float*)d_out);
}
